// EncoderBlock_66451734004044
// MI455X (gfx1250) — compile-verified, matrix operand order checked by reading, not run on hardware
//
#include <hip/hip_runtime.h>

typedef __attribute__((ext_vector_type(16))) __bf16 v16bf;
typedef __attribute__((ext_vector_type(8)))  float  v8f;

#define INV_KP_EXTENT 1.7320508075688772f   // 1/KP_EXTENT, KP_EXTENT = 1/sqrt(3)

// ---------------- workspace layout (bytes) ----------------
// kpw_frag : [8 nt][108 c][32 lane][16 h]  bf16  = 884736 B
// wcat_frag: [16 nt][8 c][32 lane][16 h]   bf16  = 131072 B
// bias_cat : [256] f32                           =   1024 B
#define WS_KPW   0
#define WS_WCAT  884736
#define WS_BIAS  (884736 + 131072)

// ---------------- dynamic LDS layout (bytes) ----------------
// feature rows padded to 528B (132 f32) so the two B-fragment lane halves
// (row offset 8) land in disjoint LDS bank groups (8*132 mod 64 == 32).
#define FB_ROWF  132               // floats per padded row
#define FB_WAVEF (32 * FB_ROWF)    // 4224 floats per wave buffer
#define OFF_NPOS 0                 // [16 q][32 nb][4] f32          = 8192
#define OFF_QPTS 8192              // [16 q][4] f32                 = 256
#define OFF_QIDX 8448              // [16] i32                      = 64
#define OFF_IND  8512              // [16 q][32 nb] i32             = 2048
#define OFF_FBUF 10560             // 8 waves * [32 nb][132] f32    = 135168
#define OFF_A2   145728            // [108 c][32][16] bf16          = 110592
#define OFF_A3   256320            // [8 c][32][16] bf16            = 8192
#define OFF_DUMMY 264512           // 32 lanes * 2B write sink      = 64
#define SMEM_BYTES 264576

__device__ __forceinline__ v8f wmma_bf16(v16bf a, v16bf b, v8f c) {
  // (neg_a, A, neg_b, B, c_mod, C, reuse_a, reuse_b)
  return __builtin_amdgcn_wmma_f32_16x16x32_bf16(false, a, false, b, (short)0, c,
                                                 false, false);
}

// Repack weights into B-fragment order so each lane's fragment is one
// contiguous 32B (16 x bf16) vector load.
__global__ void prep_kernel(const float* __restrict__ kpw,
                            const float* __restrict__ lin_w,
                            const float* __restrict__ lin_b,
                            const float* __restrict__ proj_w,
                            const float* __restrict__ proj_b,
                            __bf16* __restrict__ kpw_frag,
                            __bf16* __restrict__ wcat_frag,
                            float* __restrict__ bias_cat)
{
  const int NKPW  = 8 * 108 * 32 * 16;   // 442368
  const int NWCAT = 16 * 8 * 32 * 16;    // 65536
  int tid = blockIdx.x * blockDim.x + threadIdx.x;
  if (tid < NKPW) {
    int e = tid;
    int h = e & 15; int lane = (e >> 4) & 31;
    int c = (e >> 9) % 108; int nt = e / (512 * 108);
    int kk = ((h >> 3) << 4) | ((lane >> 4) << 3) | (h & 7);  // K row in 32-chunk
    int flat = c * 32 + kk;                 // flattened (k, i), < 3456
    int k = flat >> 7; int i = flat & 127;
    int o = nt * 16 + (lane & 15);
    kpw_frag[e] = (__bf16)kpw[(k * 128 + i) * 128 + o];
  } else if (tid < NKPW + NWCAT) {
    int e = tid - NKPW;
    int h = e & 15; int lane = (e >> 4) & 31;
    int c = (e >> 9) & 7; int nt = e >> 12;
    int kk = ((h >> 3) << 4) | ((lane >> 4) << 3) | (h & 7);
    int r = c * 32 + kk;                    // concat row: fx(0..127) | qfeat(128..255)
    int o = nt * 16 + (lane & 15);
    float v = (r < 128) ? lin_w[r * 256 + o] : proj_w[(r - 128) * 256 + o];
    wcat_frag[e] = (__bf16)v;
  } else if (tid < NKPW + NWCAT + 256) {
    int o = tid - NKPW - NWCAT;
    bias_cat[o] = lin_b[o] + proj_b[o];
  }
}

__launch_bounds__(256)
__global__ void kpconv_block_kernel(const float* __restrict__ src_pts,
                                    const float* __restrict__ src_feats,
                                    const int* __restrict__ query_idx,
                                    const int* __restrict__ neighb_idx,
                                    const __bf16* __restrict__ kpw_frag,
                                    const __bf16* __restrict__ wcat_frag,
                                    const float* __restrict__ bias_cat,
                                    float* __restrict__ out,
                                    int n_src, int n_q)
{
  extern __shared__ char smem[];
  float*  lds_npos = (float*) (smem + OFF_NPOS);
  float*  lds_qpts = (float*) (smem + OFF_QPTS);
  int*    lds_qidx = (int*)   (smem + OFF_QIDX);
  int*    lds_ind  = (int*)   (smem + OFF_IND);
  float*  fbuf32   = (float*) (smem + OFF_FBUF);
  __bf16* a3frag   = (__bf16*)(smem + OFF_A3);

  const int t      = threadIdx.x;
  const int w      = t >> 5;          // wave id 0..7 (wave32)
  const int lane   = t & 31;
  const int laneHi = lane >> 4;       // 0: lanes 0-15, 1: lanes 16-31
  const int qbase  = blockIdx.x * 16;
  const size_t OUTB = (size_t)n_q * 3;   // q_pts block precedes features in d_out

  // ---- stage 0: query points / indices --------------------------------
  if (t < 16) {
    int gq = qbase + t;
    int cq = (gq < n_q) ? gq : (n_q - 1);
    int qi = query_idx[cq];
    lds_qidx[t] = qi;
    float px = src_pts[(size_t)qi * 3 + 0];
    float py = src_pts[(size_t)qi * 3 + 1];
    float pz = src_pts[(size_t)qi * 3 + 2];
    lds_qpts[t * 4 + 0] = px;
    lds_qpts[t * 4 + 1] = py;
    lds_qpts[t * 4 + 2] = pz;
    if (gq < n_q) {
      out[(size_t)gq * 3 + 0] = px;
      out[(size_t)gq * 3 + 1] = py;
      out[(size_t)gq * 3 + 2] = pz;
    }
  }
  __syncthreads();

  // neighbor indices + relative positions. Shadow neighbors keep npos=1e6 so
  // their kernel-point weights clip to exactly 0 -> gathered features are
  // multiplied by 0.0, so we may safely clamp shadow rows to source row 0.
  for (int p = t; p < 512; p += 256) {
    int q = p >> 5, nb = p & 31;
    int gq = qbase + q;
    int cq = (gq < n_q) ? gq : (n_q - 1);
    int raw = neighb_idx[(size_t)cq * 32 + nb];
    int ind = raw % (n_src + 1);
    bool shadow = (ind == n_src);
    float sx, sy, sz;
    if (shadow) { sx = 1e6f; sy = 1e6f; sz = 1e6f; }
    else {
      sx = src_pts[(size_t)ind * 3 + 0];
      sy = src_pts[(size_t)ind * 3 + 1];
      sz = src_pts[(size_t)ind * 3 + 2];
    }
    lds_npos[(q * 32 + nb) * 4 + 0] = sx - lds_qpts[q * 4 + 0];
    lds_npos[(q * 32 + nb) * 4 + 1] = sy - lds_qpts[q * 4 + 1];
    lds_npos[(q * 32 + nb) * 4 + 2] = sz - lds_qpts[q * 4 + 2];
    lds_ind[q * 32 + nb] = shadow ? 0 : ind;
  }

  // ---- stage 0.5: query features -> stage-3 A-fragment chunks 4..7 ----
  {
    int q = t >> 4;
    int qi = lds_qidx[q];
    int seg = t & 15;
    #pragma unroll
    for (int j = 0; j < 8; ++j) {
      int i = seg * 8 + j;
      float f = src_feats[(size_t)qi * 128 + i];
      int c3 = 4 + (i >> 5);
      int kk = i & 31;
      int hi = (kk >> 3) & 1;
      int h  = ((kk >> 4) << 3) | (kk & 7);
      a3frag[(c3 * 32 + q + hi * 16) * 16 + h] = (__bf16)f;
    }
  }
  __syncthreads();

  // ---- stage 1: per-query  WF_q[27x128] = all_w^T[27x32] x F_q[32x128] ----
  // Lane gathers its neighbor row with async DMA to LDS, the wave computes
  // kernel-point weights while the DMA is in flight, then waits on ASYNCcnt.
  for (int qq = 0; qq < 2; ++qq) {
    int q = w * 2 + qq;
    {
      int ind = lds_ind[q * 32 + lane];              // clamped (0 for shadow)
      unsigned long long gp =
          (unsigned long long)(const void*)(src_feats + (size_t)ind * 128);
      unsigned ldsa =
          (unsigned)(uintptr_t)(smem + OFF_FBUF + (w * FB_WAVEF + lane * FB_ROWF) * 4);
      #pragma unroll
      for (int s = 0; s < 32; ++s) {
        // copies 16B per lane: 512B/row across the wave per instruction
        asm volatile("global_load_async_to_lds_b128 %0, %1, off offset:%2"
                     :: "v"(ldsa), "v"(gp), "n"(s * 16) : "memory");
      }
    }

    // overlap with DMA: load the 16 neighbor positions this lane needs once,
    // then build both A fragments (kernel-point weights) from registers.
    float nx[16], ny[16], nz[16];
    #pragma unroll
    for (int h = 0; h < 16; ++h) {
      int nb = ((h >> 3) << 4) | (laneHi << 3) | (h & 7);
      const float* np = &lds_npos[(q * 32 + nb) * 4];
      nx[h] = np[0]; ny[h] = np[1]; nz[h] = np[2];
    }
    v16bf afr[2];
    #pragma unroll
    for (int m = 0; m < 2; ++m) {
      int k = m * 16 + (lane & 15);            // kernel-point row this lane owns
      bool kv = (k < 27);
      float kx = 0.f, ky = 0.f, kz = 0.f;
      if (kv) {
        int ka = k / 9; int kr_ = k - ka * 9;
        int kb = kr_ / 3; int kc = kr_ - kb * 3;
        kx = (float)(ka - 1); ky = (float)(kb - 1); kz = (float)(kc - 1);
      }
      #pragma unroll
      for (int h = 0; h < 16; ++h) {
        float dx = nx[h] - kx, dy = ny[h] - ky, dz = nz[h] - kz;
        float s  = dx * dx + dy * dy + dz * dz;
        float wv = 1.0f - __builtin_amdgcn_sqrtf(s) * INV_KP_EXTENT;
        wv = kv ? fmaxf(wv, 0.0f) : 0.0f;
        afr[m][h] = (__bf16)wv;
      }
    }

    asm volatile("s_wait_asynccnt 0" ::: "memory");  // DMA complete

    #pragma unroll
    for (int nt = 0; nt < 8; ++nt) {
      // B fragment from the async-gathered f32 rows
      v16bf bfrag;
      int ic = nt * 16 + (lane & 15);
      #pragma unroll
      for (int h = 0; h < 16; ++h) {
        int nb = ((h >> 3) << 4) | (laneHi << 3) | (h & 7);
        bfrag[h] = (__bf16)fbuf32[w * FB_WAVEF + nb * FB_ROWF + ic];
      }
      #pragma unroll
      for (int m = 0; m < 2; ++m) {
        v8f c = {};
        c = wmma_bf16(afr[m], bfrag, c);
        #pragma unroll
        for (int v = 0; v < 8; ++v) {
          int kr = m * 16 + v + laneHi * 8;    // C row = vgpr + 8*(lane>=16)
          // rows >= 27 are padding: redirect the store to a per-lane write
          // sink instead of exec-masking (v_cndmask, no divergent branch)
          int flat = kr * 128 + ic;            // flattened (k,i) position
          int c2 = flat >> 5; int kk = flat & 31;
          int hi = (kk >> 3) & 1;
          int h2 = ((kk >> 4) << 3) | (kk & 7);
          int off = (kr < 27)
                  ? (OFF_A2 + ((c2 * 32 + q + hi * 16) * 16 + h2) * 2)
                  : (OFF_DUMMY + lane * 2);
          *(__bf16*)(smem + off) = (__bf16)c[v];
        }
      }
    }
  }
  __syncthreads();

  // ---- stage 2: fx[16x128] = WF[16x3456] x kpw[3456x128], leaky-ReLU ----
  {
    int nt2 = w;                               // each wave owns one mid N-tile
    v8f acc = {};
    #pragma unroll 4
    for (int c = 0; c < 108; ++c) {
      int pc = (c + 16 < 108) ? (c + 16) : 107;  // stream-ahead hint into GL2
      __builtin_prefetch(&kpw_frag[(size_t)((nt2 * 108 + pc) * 32 + lane) * 16], 0, 0);
      v16bf A = *(const v16bf*)(smem + OFF_A2 + ((c * 32 + lane) * 16) * 2);
      v16bf B = *(const v16bf*)&kpw_frag[(size_t)((nt2 * 108 + c) * 32 + lane) * 16];
      acc = wmma_bf16(A, B, acc);
    }
    int o  = nt2 * 16 + (lane & 15);
    int c3 = o >> 5; int kk = o & 31;
    int hi = (kk >> 3) & 1;
    int h  = ((kk >> 4) << 3) | (kk & 7);
    #pragma unroll
    for (int v = 0; v < 8; ++v) {
      float f = acc[v];
      f = (f > 0.0f) ? f : 0.01f * f;          // leaky_relu(0.01)
      int q = v + laneHi * 8;
      a3frag[(c3 * 32 + q + hi * 16) * 16 + h] = (__bf16)f;
    }
  }
  __syncthreads();

  // ---- stage 3: out[16x256] = [fx | qfeat][16x256] x wcat[256x256] + bias ----
  #pragma unroll
  for (int t3 = 0; t3 < 2; ++t3) {
    int nt3 = w * 2 + t3;                      // 16 out N-tiles over 8 waves
    v8f acc = {};
    #pragma unroll
    for (int c = 0; c < 8; ++c) {
      v16bf A = *(const v16bf*)&a3frag[(c * 32 + lane) * 16];
      v16bf B = *(const v16bf*)&wcat_frag[(size_t)((nt3 * 8 + c) * 32 + lane) * 16];
      acc = wmma_bf16(A, B, acc);
    }
    int o = nt3 * 16 + (lane & 15);
    float bb = bias_cat[o];
    #pragma unroll
    for (int v = 0; v < 8; ++v) {
      int q  = v + laneHi * 8;
      int gq = qbase + q;
      if (gq < n_q) out[OUTB + (size_t)gq * 256 + o] = acc[v] + bb;
    }
  }
}

extern "C" void kernel_launch(void* const* d_in, const int* in_sizes, int n_in,
                              void* d_out, int out_size, void* d_ws, size_t ws_size,
                              hipStream_t stream) {
  const float* src_pts   = (const float*)d_in[0];
  const float* src_feats = (const float*)d_in[1];
  const float* kpw       = (const float*)d_in[2];
  const float* lin_w     = (const float*)d_in[3];
  const float* lin_b     = (const float*)d_in[4];
  const float* proj_w    = (const float*)d_in[5];
  const float* proj_b    = (const float*)d_in[6];
  const int* query_idx   = (const int*)d_in[7];
  const int* neighb_idx  = (const int*)d_in[8];
  int n_src = in_sizes[0] / 3;
  int n_q   = in_sizes[7];
  float* out = (float*)d_out;

  char* ws = (char*)d_ws;
  __bf16* kpw_frag  = (__bf16*)(ws + WS_KPW);
  __bf16* wcat_frag = (__bf16*)(ws + WS_WCAT);
  float*  bias_cat  = (float*) (ws + WS_BIAS);

  (void)hipFuncSetAttribute((const void*)kpconv_block_kernel,
                            hipFuncAttributeMaxDynamicSharedMemorySize, SMEM_BYTES);

  int preptot = 8 * 108 * 32 * 16 + 16 * 8 * 32 * 16 + 256;
  prep_kernel<<<(preptot + 255) / 256, 256, 0, stream>>>(
      kpw, lin_w, lin_b, proj_w, proj_b, kpw_frag, wcat_frag, bias_cat);

  int nblocks = (n_q + 15) / 16;
  kpconv_block_kernel<<<nblocks, 256, SMEM_BYTES, stream>>>(
      src_pts, src_feats, query_idx, neighb_idx,
      kpw_frag, wcat_frag, bias_cat, out, n_src, n_q);
}
